// Attention_33268816674873
// MI455X (gfx1250) — matrix operand order repaired and checked by reading, not compiled, not run on hardware
//
#include <hip/hip_runtime.h>

// ---------------------------------------------------------------------------
// Coverage attention for MI455X (gfx1250, wave32, WMMA).
//   B=8, T=64, S=256, H=768
// Pipeline:
//   1) split fp32 -> bf16 hi/lo (and transpose W matrices)
//   2) enc_feature = enc @ W_h^T   via v_wmma_f32_16x16x32_bf16 (3-product split)
//   3) dec_fea     = dec @ W_dec^T + b_dec (same WMMA kernel, bias fused)
//   4) sequential scan over T (1 workgroup per batch, cov/softmax in LDS,
//      tanh/exp on the TRANS pipe via v_exp_f32)
//   5) ht_hat = attn @ enc  batched WMMA GEMM (serial dep removed from scan)
// GEMM: each wave owns a 16x64 C tile (4 N-subtiles) so the A fragment is
// reused across 4 WMMA columns -> 12 WMMAs per 4 b128 A loads + 128 16-bit B loads per K-slab.
// ---------------------------------------------------------------------------

typedef __attribute__((ext_vector_type(16))) __bf16 v16bf;
typedef __attribute__((ext_vector_type(8)))  __bf16 v8bf;
typedef __attribute__((ext_vector_type(8)))  float  v8f;

#define HDIM 768
#define BATCH 8
#define TLEN 64
#define SLEN 256
#define NTILE 4            // N-subtiles (16 cols each) per wave
#define LOG2E 1.44269504088896f

static __device__ __forceinline__ __bf16 f2bf(float f) {
  unsigned u = __builtin_bit_cast(unsigned, f);
  u += 0x7FFFu + ((u >> 16) & 1u);            // round-to-nearest-even
  unsigned short s = (unsigned short)(u >> 16);
  return __builtin_bit_cast(__bf16, s);
}
static __device__ __forceinline__ float bf2f(__bf16 h) {
  unsigned u = ((unsigned)__builtin_bit_cast(unsigned short, h)) << 16;
  return __builtin_bit_cast(float, u);
}

// ---------------------------------------------------------------------------
// fp32 -> bf16 (hi) + bf16 (residual lo)
// ---------------------------------------------------------------------------
__global__ void split_bf16_kernel(const float* __restrict__ src,
                                  __bf16* __restrict__ hi,
                                  __bf16* __restrict__ lo, int n) {
  int i = blockIdx.x * blockDim.x + threadIdx.x;
  if (i < n) {
    float x = src[i];
    __bf16 h = f2bf(x);
    hi[i] = h;
    lo[i] = f2bf(x - bf2f(h));
  }
}

// out[k][c] = W[c][k]  (so GEMM reads B row-major [K,N]); split into hi/lo.
__global__ void transpose_split_kernel(const float* __restrict__ W,
                                       __bf16* __restrict__ hi,
                                       __bf16* __restrict__ lo, int n) {
  int i = blockIdx.x * blockDim.x + threadIdx.x;
  if (i < n * n) {
    int k = i / n, c = i % n;
    float x = W[c * n + k];
    __bf16 h = f2bf(x);
    hi[(size_t)k * n + c] = h;
    lo[(size_t)k * n + c] = f2bf(x - bf2f(h));
  }
}

// ---------------------------------------------------------------------------
// C[M,N] = A[M,K] @ B[K,N] (+bias[N]) in near-fp32 via 3 bf16 WMMA products.
// One wave computes a 16x(16*NTILE) C tile; A fragment reused across the
// NTILE columns. Batched via blockIdx.z. Tile loads follow the CDNA5 16-bit
// WMMA VGPR layouts (ISA 7.12.2): A lane<->M regs<->K, B lane<->N regs<->K.
// ---------------------------------------------------------------------------
__global__ __launch_bounds__(32) void gemm_bf16split_kernel(
    const __bf16* __restrict__ Ah, const __bf16* __restrict__ Al,
    const __bf16* __restrict__ Bh, const __bf16* __restrict__ Bl,
    float* __restrict__ C, const float* __restrict__ bias,
    int M, int N, int K,
    long long sA, long long sB, long long sC) {
  const int lane = threadIdx.x;
  const int n0 = blockIdx.x * (16 * NTILE);
  const int m0 = blockIdx.y * 16;
  const __bf16* ah = Ah + (long long)blockIdx.z * sA;
  const __bf16* al = Al + (long long)blockIdx.z * sA;
  const __bf16* bh = Bh + (long long)blockIdx.z * sB;
  const __bf16* bl = Bl + (long long)blockIdx.z * sB;
  float* cc = C + (long long)blockIdx.z * sC;

  const int mrow = m0 + (lane & 15);          // A: lane <-> M
  const int ko   = (lane >> 4) * 8;           // A: low lanes K 0-7/16-23, high 8-15/24-31
  const int kbl  = (lane >> 4) * 8;           // B: lane <-> N, regs <-> K (same K order as A)

  v8f acc[NTILE] = {};

  for (int kb = 0; kb < K; kb += 32) {
    const __bf16* pa  = ah + (long long)mrow * K + kb + ko;
    const __bf16* pal = al + (long long)mrow * K + kb + ko;
    const __bf16* pbrow  = bh + (long long)(kb + kbl) * N + n0 + (lane & 15);
    const __bf16* pblrow = bl + (long long)(kb + kbl) * N + n0 + (lane & 15);

    if (kb + 32 < K) {                        // global_prefetch_b8, near scope
      __builtin_prefetch(pa + 32, 0, 3);
      __builtin_prefetch(pbrow + (size_t)32 * N, 0, 3);
    }

    v8bf a0h = *(const v8bf*)(pa);
    v8bf a1h = *(const v8bf*)(pa + 16);
    v16bf avh = __builtin_shufflevector(a0h, a1h, 0,1,2,3,4,5,6,7,8,9,10,11,12,13,14,15);
    v8bf a0l = *(const v8bf*)(pal);
    v8bf a1l = *(const v8bf*)(pal + 16);
    v16bf avl = __builtin_shufflevector(a0l, a1l, 0,1,2,3,4,5,6,7,8,9,10,11,12,13,14,15);

#pragma unroll
    for (int nt = 0; nt < NTILE; ++nt) {
      v16bf bvh, bvl;
#pragma unroll
      for (int s = 0; s < 16; ++s) {
        const int kr = s < 8 ? s : s + 8;
        bvh[s] = pbrow[(long long)kr * N + nt * 16];
        bvl[s] = pblrow[(long long)kr * N + nt * 16];
      }
      acc[nt] = __builtin_amdgcn_wmma_f32_16x16x32_bf16(false, avh, false, bvh, (short)0, acc[nt], false, false);
      acc[nt] = __builtin_amdgcn_wmma_f32_16x16x32_bf16(false, avh, false, bvl, (short)0, acc[nt], false, false);
      acc[nt] = __builtin_amdgcn_wmma_f32_16x16x32_bf16(false, avl, false, bvh, (short)0, acc[nt], false, false);
    }
  }

  const int rbase = m0 + ((lane >> 4) << 3);  // C: lanes 16-31 hold M+8
#pragma unroll
  for (int nt = 0; nt < NTILE; ++nt) {
    const int col = n0 + nt * 16 + (lane & 15);
    const float bv = bias ? bias[col] : 0.f;
#pragma unroll
    for (int r = 0; r < 8; ++r)
      cc[(long long)(rbase + r) * N + col] = acc[nt][r] + bv;
  }
}

// ---------------------------------------------------------------------------
// init: denom = sum(decoder_input_mask); zero the loss accumulator
// ---------------------------------------------------------------------------
__global__ void init_denom_kernel(const float* __restrict__ dmask,
                                  float* __restrict__ scal) {
  __shared__ float r[BATCH * TLEN];
  int t = threadIdx.x;
  r[t] = dmask[t];
  __syncthreads();
  for (int o = (BATCH * TLEN) / 2; o > 0; o >>= 1) {
    if (t < o) r[t] += r[t + o];
    __syncthreads();
  }
  if (t == 0) { scal[1] = r[0]; scal[0] = 0.f; }
}

// ---------------------------------------------------------------------------
// Sequential coverage scan: one workgroup (1024 thr = 32 waves) per batch.
// Per step: 256 score rows (8 per wave, 768-long tanh-dot on TRANS pipe),
// LDS softmax over S, coverage/loss update, attn stored fp32 + bf16 hi/lo.
// ---------------------------------------------------------------------------
__global__ __launch_bounds__(1024) void scan_kernel(
    const float* __restrict__ encf, const float* __restrict__ decf,
    const float* __restrict__ wc, const float* __restrict__ vv,
    const float* __restrict__ pmask, const float* __restrict__ dmask,
    const float* __restrict__ cov0,
    float* __restrict__ attn_out, float* __restrict__ cov_out,
    __bf16* __restrict__ attn_hi, __bf16* __restrict__ attn_lo,
    float* __restrict__ scal) {
  __shared__ float cov[SLEN];
  __shared__ float sc[SLEN];
  __shared__ float dect[HDIM];
  __shared__ float red[SLEN];
  __shared__ float lossacc;

  const int b = blockIdx.x;
  const int tid = threadIdx.x;
  const int lane = tid & 31, wave = tid >> 5;

  for (int s = tid; s < SLEN; s += 1024) cov[s] = cov0[b * SLEN + s];
  if (tid == 0) lossacc = 0.f;
  __syncthreads();

  for (int t = 0; t < TLEN; ++t) {
    for (int h = tid; h < HDIM; h += 1024)
      dect[h] = decf[((size_t)(b * TLEN + t)) * HDIM + h];
    __syncthreads();

    // scores[s] = sum_h v[h] * tanh(encf + dec_t + cov*w_c)
    for (int i = 0; i < 8; ++i) {
      int s = wave * 8 + i;
      float cv = cov[s];
      const float* er = encf + ((size_t)(b * SLEN + s)) * HDIM;
      float acc = 0.f;
      for (int h = lane; h < HDIM; h += 32) {
        float x = er[h] + dect[h] + cv * wc[h];
        float e2 = __builtin_amdgcn_exp2f(x * (2.f * LOG2E));   // v_exp_f32
        acc += vv[h] * (1.f - 2.f / (1.f + e2));                // tanh(x)
      }
      for (int o = 16; o > 0; o >>= 1) acc += __shfl_xor(acc, o, 32);
      if (lane == 0) sc[s] = acc;
    }
    __syncthreads();

    float val = (tid < SLEN) ? sc[tid] : 0.f;
    if (tid < SLEN) red[tid] = val;
    __syncthreads();
    for (int o = SLEN / 2; o > 0; o >>= 1) {
      if (tid < o) red[tid] = fmaxf(red[tid], red[tid + o]);
      __syncthreads();
    }
    float mx = red[0];
    __syncthreads();

    float p = 0.f;
    if (tid < SLEN) { p = __builtin_amdgcn_exp2f((val - mx) * LOG2E); red[tid] = p; }
    __syncthreads();
    for (int o = SLEN / 2; o > 0; o >>= 1) {
      if (tid < o) red[tid] += red[tid + o];
      __syncthreads();
    }
    float Z = red[0];
    __syncthreads();

    float am = 0.f;
    if (tid < SLEN) { am = (p / Z) * pmask[b * SLEN + tid]; red[tid] = am; }
    __syncthreads();
    for (int o = SLEN / 2; o > 0; o >>= 1) {
      if (tid < o) red[tid] += red[tid + o];
      __syncthreads();
    }
    float msum = red[0];
    __syncthreads();

    if (tid < SLEN) {
      float attn = am / msum;
      float cv = cov[tid];
      red[tid] = fminf(attn, cv);                 // coverage-loss term
      cov[tid] = cv + attn;
      size_t o = ((size_t)(b * TLEN + t)) * SLEN + tid;
      attn_out[o] = attn;
      __bf16 hh = f2bf(attn);
      attn_hi[o] = hh;
      attn_lo[o] = f2bf(attn - bf2f(hh));
    }
    __syncthreads();
    for (int o = SLEN / 2; o > 0; o >>= 1) {
      if (tid < o) red[tid] += red[tid + o];
      __syncthreads();
    }
    if (tid == 0) lossacc += red[0] * dmask[b * TLEN + t];
    __syncthreads();
  }

  if (tid < SLEN) cov_out[b * SLEN + tid] = cov[tid];
  __syncthreads();
  if (tid == 0) atomicAdd(&scal[0], lossacc);
}

__global__ void finalize_kernel(const float* __restrict__ scal,
                                float* __restrict__ out_loss) {
  out_loss[0] = scal[0] / scal[1];
}

// ---------------------------------------------------------------------------
extern "C" void kernel_launch(void* const* d_in, const int* in_sizes, int n_in,
                              void* d_out, int out_size, void* d_ws, size_t ws_size,
                              hipStream_t stream) {
  (void)in_sizes; (void)n_in; (void)out_size; (void)ws_size;
  const float* dec   = (const float*)d_in[0];   // [B,T,H]
  const float* dmask = (const float*)d_in[1];   // [B,T]
  const float* enc   = (const float*)d_in[2];   // [B,S,H]
  const float* pmask = (const float*)d_in[3];   // [B,S]
  const float* cov0  = (const float*)d_in[4];   // [B,S]
  const float* Wh    = (const float*)d_in[5];   // [H,H]
  const float* Wd    = (const float*)d_in[6];   // [H,H]
  const float* bdec  = (const float*)d_in[7];   // [H]
  const float* wc    = (const float*)d_in[8];   // [H]
  const float* vv    = (const float*)d_in[9];   // [H]

  const size_t nEnc = (size_t)BATCH * SLEN * HDIM;   // 1572864
  const size_t nDec = (size_t)BATCH * TLEN * HDIM;   // 393216
  const size_t nW   = (size_t)HDIM * HDIM;           // 589824
  const size_t nAtt = (size_t)BATCH * TLEN * SLEN;   // 131072

  char* ws = (char*)d_ws;
  size_t off = 0;
  auto alloc = [&](size_t bytes) {
    size_t o = off;
    off += (bytes + 255) & ~(size_t)255;
    return o;
  };
  __bf16* enc_hi = (__bf16*)(ws + alloc(nEnc * 2));
  __bf16* enc_lo = (__bf16*)(ws + alloc(nEnc * 2));
  __bf16* dec_hi = (__bf16*)(ws + alloc(nDec * 2));
  __bf16* dec_lo = (__bf16*)(ws + alloc(nDec * 2));
  __bf16* wh_hi  = (__bf16*)(ws + alloc(nW * 2));
  __bf16* wh_lo  = (__bf16*)(ws + alloc(nW * 2));
  __bf16* wd_hi  = (__bf16*)(ws + alloc(nW * 2));
  __bf16* wd_lo  = (__bf16*)(ws + alloc(nW * 2));
  float*  encf   = (float*)(ws + alloc(nEnc * 4));
  float*  decf   = (float*)(ws + alloc(nDec * 4));
  __bf16* att_hi = (__bf16*)(ws + alloc(nAtt * 2));
  __bf16* att_lo = (__bf16*)(ws + alloc(nAtt * 2));
  float*  scal   = (float*)(ws + alloc(8));        // [0]=loss num, [1]=denom

  // d_out: ht_hat [B,T,H] | attn_dist [B,T,S] | loss [1] | coverage [B,S]
  float* out_ht   = (float*)d_out;
  float* out_attn = out_ht + nDec;
  float* out_loss = out_attn + nAtt;
  float* out_cov  = out_loss + 1;

  const int thr = 256;
  split_bf16_kernel<<<(unsigned)((nEnc + thr - 1) / thr), thr, 0, stream>>>(enc, enc_hi, enc_lo, (int)nEnc);
  split_bf16_kernel<<<(unsigned)((nDec + thr - 1) / thr), thr, 0, stream>>>(dec, dec_hi, dec_lo, (int)nDec);
  transpose_split_kernel<<<(unsigned)((nW + thr - 1) / thr), thr, 0, stream>>>(Wh, wh_hi, wh_lo, HDIM);
  transpose_split_kernel<<<(unsigned)((nW + thr - 1) / thr), thr, 0, stream>>>(Wd, wd_hi, wd_lo, HDIM);

  // enc_feature = enc @ Wh^T   (M=2048, N=768, K=768)
  dim3 g1(HDIM / (16 * NTILE), (BATCH * SLEN) / 16, 1);
  gemm_bf16split_kernel<<<g1, 32, 0, stream>>>(enc_hi, enc_lo, wh_hi, wh_lo, encf,
                                               nullptr, BATCH * SLEN, HDIM, HDIM, 0, 0, 0);
  // dec_fea = dec @ Wd^T + b   (M=512, N=768, K=768)
  dim3 g2(HDIM / (16 * NTILE), (BATCH * TLEN) / 16, 1);
  gemm_bf16split_kernel<<<g2, 32, 0, stream>>>(dec_hi, dec_lo, wd_hi, wd_lo, decf,
                                               bdec, BATCH * TLEN, HDIM, HDIM, 0, 0, 0);

  init_denom_kernel<<<1, BATCH * TLEN, 0, stream>>>(dmask, scal);

  scan_kernel<<<BATCH, 1024, 0, stream>>>(encf, decf, wc, vv, pmask, dmask, cov0,
                                          out_attn, out_cov, att_hi, att_lo, scal);

  // ht_hat[b] = attn[b] @ enc[b]   (batched: M=64, N=768, K=256)
  dim3 g3(HDIM / (16 * NTILE), TLEN / 16, BATCH);
  gemm_bf16split_kernel<<<g3, 32, 0, stream>>>(att_hi, att_lo, enc_hi, enc_lo, out_ht,
                                               nullptr, TLEN, HDIM, SLEN,
                                               (long long)TLEN * SLEN,
                                               (long long)SLEN * HDIM,
                                               (long long)TLEN * HDIM);

  finalize_kernel<<<1, 1, 0, stream>>>(scal, out_loss);
}
